// MixedRepeatHeads_12481174962961
// MI455X (gfx1250) — matrix operand order repaired and checked by reading, not compiled, not run on hardware
//
#include <hip/hip_runtime.h>
#include <hip/hip_bf16.h>

// MixedRepeatHeads on gfx1250 (MI455X, wave32).
// Pipeline:
//   cvt     : proj_w, out_w f32 -> bf16 (once; removes cvt from GEMM hot loop)
//   GEMM1 (WMMA bf16): projT[b,n,s] = sum_e x[b,e,t=s] * proj_w[n,e] + proj_b[n]
//   mixer = exponential-decay prefix scan (chunked, 3 passes)  -> hidden[b,s,n] (bf16)
//   GEMM2 (WMMA bf16): out[b,e,s] = sum_n hidden[b,s,n] * out_w[e,n] + out_b[e]
// B tiles: double-buffered in LDS, filled by CDNA5 async copies
// (global_load_async_to_lds_b128, ASYNCcnt) issued one k-step ahead;
// s_wait_asynccnt 2 drains only the older stage. A operands software-prefetched
// into registers one k-step ahead. B fragments double-buffered against WMMA.
// Roofline: scan removes the 137-GFLOP S*S mixer matmul; total HBM traffic
// ~230 MB -> ~10 us at 23.3 TB/s; remaining 2x34 GFLOP GEMMs ride bf16 WMMA.
// Workspace layout (~73 MB):
//   projT  : bf16 [8][1024][2048]  (33.6 MB)
//   hidden : bf16 [8][2048][1024]  (33.6 MB)
//   carry  : f32  [8][8][32][128]  ( 1.0 MB)
//   wprojb : bf16 [1024][1024]     ( 2.1 MB)
//   woutb  : bf16 [1024][1024]     ( 2.1 MB)

#define BATCH 8
#define DIM   1024
#define SEQ   2048
#define HEADS 8
#define HID   128
#define NCHUNK 32
#define CHUNK 64

typedef __attribute__((ext_vector_type(16))) __bf16 v16bf;
typedef __attribute__((ext_vector_type(8)))  __bf16 v8bf;
typedef __attribute__((ext_vector_type(4)))  __bf16 v4bf;
typedef __attribute__((ext_vector_type(2)))  __bf16 v2bf;
typedef __attribute__((ext_vector_type(8)))  float  v8f;

// ---------------------------------------------------------------------------
// f32 -> bf16 weight conversion (1024x1024), one float4 per thread.
// ---------------------------------------------------------------------------
__global__ __launch_bounds__(256) void cvt_w_bf16(
    const float* __restrict__ src, __bf16* __restrict__ dst)
{
  const int i = (blockIdx.x * 256 + threadIdx.x) * 4;
  float4 v = *(const float4*)(src + i);
  v4bf o;
  o[0] = (__bf16)v.x; o[1] = (__bf16)v.y; o[2] = (__bf16)v.z; o[3] = (__bf16)v.w;
  *(v4bf*)(dst + i) = o;
}

// ---------------------------------------------------------------------------
// Shared GEMM: C(16384x1024) = A(16384x1024) * W^T + bias, K = 1024.
// A_FROM_X = true : A[t,e] = x[b,e,t] (f32, transposed) -> reg-prefetched, packed
//                   v2bf -> LDS; C stored transposed as bf16 projT[b,n,s].
// A_FROM_X = false: A[s,k] = hidden bf16 row-major -> reg-prefetched fragments;
//                   C stored transposed as f32 out[b,n,s].
// Workgroup: 256 threads = 8 waves; tile 128(s) x 128(n); k-step 32.
// ---------------------------------------------------------------------------
template<bool A_FROM_X>
__global__ __launch_bounds__(256) void gemm_bf16_wmma(
    const float* __restrict__ Ax,     // x, used when A_FROM_X
    const __bf16* __restrict__ Abf,   // hidden, used when !A_FROM_X
    const __bf16* __restrict__ Wbf,   // (1024,1024) bf16 row-major: W[n][k]
    const float* __restrict__ bias,   // (1024)
    __bf16* __restrict__ Cbf,         // projT, used when A_FROM_X
    float* __restrict__ Cf)           // out, used when !A_FROM_X
{
  constexpr int LDT   = 40;     // LDS row stride (elems): 80B, conflict-free b128 reads
  constexpr int KSTEP = 32;
  __shared__ __bf16 lsB[2][128 * LDT];          // double-buffered async target
  __shared__ __bf16 lsA[A_FROM_X ? 128 * LDT : 1];

  const int tid  = threadIdx.x;
  const int lane = tid & 31;
  const int wv   = tid >> 5;       // wave id 0..7
  const int lh   = lane >> 4;      // lane half
  const int l15  = lane & 15;
  const int s0   = blockIdx.x * 128;
  const int n0   = blockIdx.y * 128;
  const int b    = blockIdx.z;

  v8f acc[8];
#pragma unroll
  for (int j = 0; j < 8; ++j)
#pragma unroll
    for (int i = 0; i < 8; ++i) acc[j][i] = 0.0f;

  const float*  xb = Ax + (size_t)b * DIM * SEQ;                       // x[b] : (E, S)
  const __bf16* hb = Abf + (size_t)b * SEQ * DIM
                         + (size_t)(s0 + wv * 16 + l15) * DIM;         // hidden row, this lane

  // async B-stage parameters (flat low 32 bits of a shared pointer == LDS byte address)
  const uint64_t wbase = (uint64_t)(uintptr_t)Wbf;
  const uint32_t ldsB0 = (uint32_t)(uintptr_t)&lsB[0][0];
  const int      c0    = tid * 2;   // 2x16B chunks per thread; 512 chunks = 128 rows x 64B

  // issue the 8KB W^T tile copy for step k0 into LDS buffer `buf`
  auto issueB = [&](int k0, int buf) {
#pragma unroll
    for (int rep = 0; rep < 2; ++rep) {
      const int c = c0 + rep;
      const int row = c >> 2;
      const int part = (c & 3) * 16;                        // 16B sub-chunk within 64B row
      uint32_t lo = ldsB0 + (uint32_t)(buf * (128 * LDT * 2) + row * (LDT * 2) + part);
      uint32_t go = (uint32_t)((((n0 + row) * DIM + k0) * 2) + part);
      asm volatile("global_load_async_to_lds_b128 %0, %1, %2"
                   :: "v"(lo), "v"(go), "s"(wbase) : "memory");
    }
  };

  // A operand register prefetch (one k-step ahead)
  float2 ax[8];                  // A_FROM_X: 16 f32 of the 128x32 x-tile
  v8bf   galo, gahi;             // !A_FROM_X: this lane's A fragment halves
  auto prefetchA = [&](int k0) {
    if constexpr (A_FROM_X) {
      int idx = tid;
#pragma unroll
      for (int rep = 0; rep < 8; ++rep, idx += 256) {
        const int tt = idx & 127;
        const int ee = (idx >> 7) * 2;
        ax[rep].x = xb[(size_t)(k0 + ee) * SEQ + s0 + tt];
        ax[rep].y = xb[(size_t)(k0 + ee + 1) * SEQ + s0 + tt];
      }
    } else {
      const __bf16* ar = hb + k0;
      galo = *(const v8bf*)(ar + lh * 8);
      gahi = *(const v8bf*)(ar + 16 + lh * 8);
    }
  };

  // ---- prologue: stage 0 in flight before the loop
  issueB(0, 0);
  prefetchA(0);

  for (int k0 = 0; k0 < DIM; k0 += KSTEP) {
    const int  cur     = (k0 >> 5) & 1;
    const bool hasNext = (k0 + KSTEP) < DIM;

    if (hasNext) issueB(k0 + KSTEP, cur ^ 1);     // next B tile under this step's compute

    v16bf af;
    if constexpr (A_FROM_X) {
      // commit prefetched x values to LDS (pack pairs along e), then prefetch next
      int idx = tid;
#pragma unroll
      for (int rep = 0; rep < 8; ++rep, idx += 256) {
        const int tt = idx & 127;
        const int ee = (idx >> 7) * 2;
        v2bf pk; pk[0] = (__bf16)ax[rep].x; pk[1] = (__bf16)ax[rep].y;
        *(v2bf*)(&lsA[tt * LDT + ee]) = pk;                   // ds_store_b32
      }
      if (hasNext) prefetchA(k0 + KSTEP);
    } else {
      // combine prefetched fragment, then prefetch next
#pragma unroll
      for (int i = 0; i < 8; ++i) { af[i] = galo[i]; af[8 + i] = gahi[i]; }
      if (hasNext) prefetchA(k0 + KSTEP);
    }

    // drain only the older async stage (in-order completion), then make LDS visible
    if (hasNext) asm volatile("s_wait_asynccnt 0x2" ::: "memory");
    else         asm volatile("s_wait_asynccnt 0x0" ::: "memory");
    __syncthreads();

    if constexpr (A_FROM_X) {
      // A fragment: lane half 0 -> K {0..7,16..23}, half 1 -> K {8..15,24..31}
      const __bf16* ar = &lsA[(wv * 16 + l15) * LDT];
      v8bf alo = *(const v8bf*)(ar + lh * 8);
      v8bf ahi = *(const v8bf*)(ar + 16 + lh * 8);
#pragma unroll
      for (int i = 0; i < 8; ++i) { af[i] = alo[i]; af[8 + i] = ahi[i]; }
    }

    // ---- 8 column subtiles, B fragment double-buffered against WMMA
    const __bf16* brow = &lsB[cur][l15 * LDT + lh * 8];
    auto loadB = [&](int ns) -> v16bf {
      const __bf16* br = brow + ns * 16 * LDT;
      v8bf b0 = *(const v8bf*)(br);
      v8bf b1 = *(const v8bf*)(br + 16);
      v16bf r;
#pragma unroll
      for (int i = 0; i < 8; ++i) { r[i] = b0[i]; r[8 + i] = b1[i]; }
      return r;
    };
    v16bf bcur = loadB(0);
#pragma unroll
    for (int ns = 0; ns < 8; ++ns) {
      v16bf bnext = bcur;
      if (ns < 7) bnext = loadB(ns + 1);      // in flight while WMMA executes
      acc[ns] = __builtin_amdgcn_wmma_f32_16x16x32_bf16(
          false, af, false, bcur, (short)0, acc[ns], false, false);
      bcur = bnext;
    }
    __syncthreads();   // protects lsA rewrite and next iteration's async target buffer
  }

  // ---- epilogue: lane holds col n = l15 (+16*ns), rows s = srow..srow+7 contiguous
  const int srow = s0 + wv * 16 + lh * 8;
#pragma unroll
  for (int ns = 0; ns < 8; ++ns) {
    const int n = n0 + ns * 16 + l15;
    const float bv = bias[n];
    if constexpr (A_FROM_X) {
      v8bf st;
#pragma unroll
      for (int i = 0; i < 8; ++i) st[i] = (__bf16)(acc[ns][i] + bv);
      *(v8bf*)(Cbf + ((size_t)b * DIM + n) * SEQ + srow) = st;   // 16B contiguous store
    } else {
      float* op = Cf + ((size_t)b * DIM + n) * SEQ + srow;
      float4 lo4 = make_float4(acc[ns][0] + bv, acc[ns][1] + bv, acc[ns][2] + bv, acc[ns][3] + bv);
      float4 hi4 = make_float4(acc[ns][4] + bv, acc[ns][5] + bv, acc[ns][6] + bv, acc[ns][7] + bv);
      *(float4*)(op)     = lo4;                                   // 32B contiguous store
      *(float4*)(op + 4) = hi4;
    }
  }
}

// ---------------------------------------------------------------------------
// Mixer = decayed prefix scan. state[s] = r*state[s-1] + beta[s]*proj[s,d];
// mixed[s] = alpha[s]*state[s] + mix_b[s].
// ColRepeat heads (h<4): beta=1,        alpha=mix_w[s]
// RowRepeat heads (h>=4): beta=mix_w[t], alpha=1
// ---------------------------------------------------------------------------
__device__ __forceinline__ float head_r(const float* decay_v, int h) {
  float d = fminf(fmaxf(decay_v[h], 0.9f), 1.0f);
  return __powf(d, 0.25f);   // d^(1/DECAY_CONST), DECAY_CONST = 4
}

__global__ __launch_bounds__(128) void mixer_pass1(
    const __bf16* __restrict__ projT, const float* __restrict__ mix_w,
    const float* __restrict__ decay_v, float* __restrict__ carry)
{
  const int chunk = blockIdx.x, h = blockIdx.y, b = blockIdx.z, d = threadIdx.x;
  __shared__ float wloc[CHUNK];
  if (threadIdx.x < CHUNK)
    wloc[threadIdx.x] = mix_w[h * SEQ + chunk * CHUNK + threadIdx.x];
  __syncthreads();
  const float r = head_r(decay_v, h);
  const bool rowrep = (h >= HEADS / 2);
  const __bf16* p = projT + ((size_t)(b * DIM + h * HID + d)) * SEQ + chunk * CHUNK;
  float st = 0.0f;
#pragma unroll
  for (int i0 = 0; i0 < CHUNK; i0 += 8) {
    v8bf pv = *(const v8bf*)(p + i0);
#pragma unroll
    for (int i = 0; i < 8; ++i) {
      float beta = rowrep ? wloc[i0 + i] : 1.0f;
      st = fmaf(beta, (float)pv[i], r * st);
    }
  }
  carry[(((size_t)b * HEADS + h) * NCHUNK + chunk) * HID + d] = st;
}

__global__ __launch_bounds__(128) void mixer_pass2(
    float* __restrict__ carry, const float* __restrict__ decay_v)
{
  const int h = blockIdx.x & (HEADS - 1), b = blockIdx.x >> 3, d = threadIdx.x;
  const float r  = head_r(decay_v, h);
  const float rc = __powf(r, (float)CHUNK);
  float* base = carry + ((size_t)b * HEADS + h) * NCHUNK * HID + d;
  float run = 0.0f;
  for (int c = 0; c < NCHUNK; ++c) {      // in-place: replace with exclusive carry-in
    float t = base[(size_t)c * HID];
    base[(size_t)c * HID] = run;
    run = fmaf(run, rc, t);
  }
}

__global__ __launch_bounds__(128) void mixer_pass3(
    const __bf16* __restrict__ projT, const float* __restrict__ mix_w,
    const float* __restrict__ mix_b, const float* __restrict__ decay_v,
    const float* __restrict__ carry, __bf16* __restrict__ hidden)
{
  const int chunk = blockIdx.x, h = blockIdx.y, b = blockIdx.z, d = threadIdx.x;
  __shared__ float wloc[CHUNK];
  __shared__ float bloc[CHUNK];
  if (threadIdx.x < CHUNK) {
    wloc[threadIdx.x] = mix_w[h * SEQ + chunk * CHUNK + threadIdx.x];
    bloc[threadIdx.x] = mix_b[h * SEQ + chunk * CHUNK + threadIdx.x];
  }
  __syncthreads();
  const float r = head_r(decay_v, h);
  const bool rowrep = (h >= HEADS / 2);
  const int n = h * HID + d;
  const __bf16* p = projT + ((size_t)(b * DIM + n)) * SEQ + chunk * CHUNK;
  float st = carry[(((size_t)b * HEADS + h) * NCHUNK + chunk) * HID + d];
  __bf16* hrow = hidden + ((size_t)b * SEQ + chunk * CHUNK) * DIM + n;
#pragma unroll
  for (int i0 = 0; i0 < CHUNK; i0 += 8) {
    v8bf pv = *(const v8bf*)(p + i0);
#pragma unroll
    for (int i = 0; i < 8; ++i) {
      float beta  = rowrep ? wloc[i0 + i] : 1.0f;
      float alpha = rowrep ? 1.0f : wloc[i0 + i];
      st = fmaf(beta, (float)pv[i], r * st);
      hrow[(size_t)(i0 + i) * DIM] = (__bf16)fmaf(alpha, st, bloc[i0 + i]);
    }
  }
}

// ---------------------------------------------------------------------------
extern "C" void kernel_launch(void* const* d_in, const int* in_sizes, int n_in,
                              void* d_out, int out_size, void* d_ws, size_t ws_size,
                              hipStream_t stream) {
  const float* x       = (const float*)d_in[0];  // (8, 1024, 2048)
  const float* proj_w  = (const float*)d_in[1];  // (8, 128, 1024) == (1024, 1024)
  const float* proj_b  = (const float*)d_in[2];  // (8, 128) == (1024)
  const float* mix_w   = (const float*)d_in[3];  // (8, 2048)
  const float* mix_b   = (const float*)d_in[4];  // (8, 2048)
  const float* decay_v = (const float*)d_in[5];  // (8)
  const float* out_w   = (const float*)d_in[6];  // (1024, 1024)
  const float* out_b   = (const float*)d_in[7];  // (1024)
  float* out = (float*)d_out;                    // (8, 1024, 2048)

  const size_t PROJ_ELEMS = (size_t)BATCH * DIM * SEQ;   // 16.7M bf16
  __bf16* projT  = (__bf16*)d_ws;
  __bf16* hidden = projT + PROJ_ELEMS;
  float*  carry  = (float*)(hidden + PROJ_ELEMS);
  __bf16* wprojb = (__bf16*)(carry + (size_t)BATCH * HEADS * NCHUNK * HID);
  __bf16* woutb  = wprojb + (size_t)DIM * DIM;           // total ws use ~73 MB

  dim3 gGemm(SEQ / 128, DIM / 128, BATCH);   // (16, 8, 8)
  dim3 gScan(NCHUNK, HEADS, BATCH);          // (32, 8, 8)
  const int cvtBlocks = (DIM * DIM) / (256 * 4);

  // 0) one-time weight conversion to bf16
  cvt_w_bf16<<<cvtBlocks, 256, 0, stream>>>(proj_w, wprojb);
  cvt_w_bf16<<<cvtBlocks, 256, 0, stream>>>(out_w, woutb);
  // 1) projection: projT[b,n,s] = x_bte @ proj_w^T + proj_b
  gemm_bf16_wmma<true><<<gGemm, 256, 0, stream>>>(x, nullptr, wprojb, proj_b, projT, nullptr);
  // 2) mixer: chunked decayed prefix scan
  mixer_pass1<<<gScan, 128, 0, stream>>>(projT, mix_w, decay_v, carry);
  mixer_pass2<<<BATCH * HEADS, 128, 0, stream>>>(carry, decay_v);
  mixer_pass3<<<gScan, 128, 0, stream>>>(projT, mix_w, mix_b, decay_v, carry, hidden);
  // 3) output projection: out[b,e,s] = hidden @ out_w^T + out_b (transposed store)
  gemm_bf16_wmma<false><<<gGemm, 256, 0, stream>>>(nullptr, hidden, woutb, out_b, nullptr, out);
}
